// DoshaGAT_9749575762551
// MI455X (gfx1250) — matrix operand order repaired and checked by reading, not compiled, not run on hardware
//
#include <hip/hip_runtime.h>
#include <hip/hip_bf16.h>

typedef _Float16 half_t;
typedef __attribute__((ext_vector_type(16))) _Float16 v16h;
typedef __attribute__((ext_vector_type(8)))  _Float16 v8h;
typedef __attribute__((ext_vector_type(8)))  float    v8f;

#define NEG_SLOPE 0.2f
#define BN_EPS    1e-5f

// ---------------- atomic float max via int/uint ordering trick ----------------
__device__ __forceinline__ void atomicMaxF(float* addr, float val) {
    if (val >= 0.0f) atomicMax((int*)addr, __float_as_int(val));
    else             atomicMin((unsigned int*)addr, __float_as_uint(val));
}

// ---------------- utility kernels ----------------
__global__ void k_fill(float* __restrict__ p, float v, int n) {
    int i = blockIdx.x * blockDim.x + threadIdx.x;
    if (i < n) p[i] = v;
}

__global__ void k_f2h(const float* __restrict__ s, half_t* __restrict__ d, int n) {
    int i = blockIdx.x * blockDim.x + threadIdx.x;
    if (i < n) d[i] = (half_t)s[i];
}

// W [K, Nout] fp32 (row-major)  ->  Wt [Npad, K] fp16 (transposed, zero pad)
__global__ void k_wtrans(const float* __restrict__ W, half_t* __restrict__ Wt,
                         int K, int Nout, int Npad) {
    int i = blockIdx.x * blockDim.x + threadIdx.x;
    int tot = Npad * K;
    if (i >= tot) return;
    int nI = i / K;
    int k  = i - nI * K;
    Wt[i] = (nI < Nout) ? (half_t)W[(size_t)k * Nout + nI] : (half_t)0.0f;
}

// ---------------- WMMA GEMM with N-register-blocking ----------------
// C[M,Nout] = A[M,K](f16) * Bt[Nout,K]^T(f16).  One wave computes a 16 x (16*NB)
// strip: a single A fragment is reused against NB B tiles per K-step, cutting
// vmem issue per WMMA from 4 loads to (2+NB)/NB loads.
// A-lane layout: lane l -> row (l&15), K = (l>=16 ? 16..31 : 0..15) contiguous.
// B-lane layout: lane l -> col (l&15), K = (l>=16 ? 16..31 : 0..15) contiguous.
template <int NB>
__global__ __launch_bounds__(256)
void k_wmma_gemm(const half_t* __restrict__ A, const half_t* __restrict__ Bt,
                 float* __restrict__ C, int M, int K, int Nout) {
    const int lane    = threadIdx.x & 31;
    const int wave    = blockIdx.x * (blockDim.x >> 5) + (threadIdx.x >> 5);
    const int groupsN = (Nout >> 4) / NB;           // N-tile groups per row
    const int tm      = (wave / groupsN) << 4;
    const int tn      = (wave % groupsN) * (16 * NB);
    if (tm >= M) return;

    const int hs = lane >> 4;     // half-select (0/1)
    const int mr = lane & 15;

    const half_t* aP = A  + (size_t)(tm + mr) * K + hs * 16;  // 16B aligned chunks
    const half_t* bP = Bt + (size_t)(tn + mr) * K + hs * 16;  // 32B aligned chunks

    v8f acc[NB];
#pragma unroll
    for (int j = 0; j < NB; ++j) acc[j] = (v8f){};

    for (int k0 = 0; k0 < K; k0 += 32) {
        v8h alo = *(const v8h*)(aP + k0);
        v8h ahi = *(const v8h*)(aP + k0 + 8);
        v16h a  = __builtin_shufflevector(alo, ahi,
                     0,1,2,3,4,5,6,7,8,9,10,11,12,13,14,15);
#pragma unroll
        for (int j = 0; j < NB; ++j) {
            v16h b = *(const v16h*)(bP + (size_t)j * 16 * K + k0);
            acc[j] = __builtin_amdgcn_wmma_f32_16x16x32_f16(
                         false, a, false, b, (short)0, acc[j], false, false);
        }
    }

    // C/D layout: VGPR r, lanes 0-15 -> row r, lanes 16-31 -> row 8+r; col = lane&15
    float* cP = C + (size_t)(tm + hs * 8) * Nout + tn + mr;
#pragma unroll
    for (int j = 0; j < NB; ++j)
#pragma unroll
        for (int r = 0; r < 8; ++r)
            cP[(size_t)r * Nout + j * 16] = acc[j][r];
}

// ---------------- attention dot products: als/ald [N, heads] ----------------
__global__ void k_attn(const float* __restrict__ Hm, int ldH, int heads, int F,
                       const float* __restrict__ as_, const float* __restrict__ ad_,
                       float* __restrict__ ALS, float* __restrict__ ALD, int Nn) {
    int i = blockIdx.x * blockDim.x + threadIdx.x;
    if (i >= Nn * heads) return;
    int n = i / heads, h = i - n * heads;
    const float* hp = Hm + (size_t)n * ldH + h * F;
    const float* sp = as_ + h * F;
    const float* dp = ad_ + h * F;
    float s = 0.f, d = 0.f;
    for (int f = 0; f < F; ++f) { float v = hp[f]; s += v * sp[f]; d += v * dp[f]; }
    ALS[i] = s; ALD[i] = d;
}

// ---------------- edge pass 1: e = lrelu(als[src]+ald[dst]); segment max ----------------
__global__ void k_edge1(const int* __restrict__ src, const int* __restrict__ dst,
                        const float* __restrict__ ALS, const float* __restrict__ ALD,
                        float* __restrict__ Ebuf, float* __restrict__ Mx,
                        int E, int heads) {
    int e = blockIdx.x * blockDim.x + threadIdx.x;
    if (e >= E) return;
    int s = src[e], d = dst[e];
    for (int h = 0; h < heads; ++h) {
        float v = ALS[(size_t)s * heads + h] + ALD[(size_t)d * heads + h];
        v = v > 0.f ? v : NEG_SLOPE * v;
        Ebuf[(size_t)e * heads + h] = v;
        atomicMaxF(&Mx[(size_t)d * heads + h], v);
    }
}

// ---------------- edge pass 2: ex = exp(e - m[dst]); segment sum ----------------
__global__ void k_edge2(const int* __restrict__ dst, const float* __restrict__ Mx,
                        float* __restrict__ Ebuf, float* __restrict__ SS,
                        int E, int heads) {
    int i = blockIdx.x * blockDim.x + threadIdx.x;
    if (i >= E * heads) return;
    int e = i / heads, h = i - e * heads;
    int d = dst[e];
    float ex = __expf(Ebuf[i] - Mx[(size_t)d * heads + h]);
    Ebuf[i] = ex;
    atomicAdd(&SS[(size_t)d * heads + h], ex);
}

// ---------------- edge pass 3: alpha = ex / (ssum[dst]+eps) ----------------
__global__ void k_alpha(const int* __restrict__ dst, const float* __restrict__ SS,
                        float* __restrict__ Ebuf, int E, int heads) {
    int i = blockIdx.x * blockDim.x + threadIdx.x;
    if (i >= E * heads) return;
    int e = i / heads, h = i - e * heads;
    int d = dst[e];
    Ebuf[i] = Ebuf[i] / (SS[(size_t)d * heads + h] + 1e-16f);
}

// ---------------- edge pass 4: AGG[dst] += h[src] * alpha (coalesced) ----------------
__global__ void k_agg(const int* __restrict__ src, const int* __restrict__ dst,
                      const float* __restrict__ Alpha, const float* __restrict__ Hm,
                      int ldH, float* __restrict__ AGG,
                      int C, int F, int heads, int total) {
    int i = blockIdx.x * blockDim.x + threadIdx.x;
    if (i >= total) return;
    int e = i / C;
    int c = i - e * C;
    int h = c / F;
    int s = src[e], d = dst[e];
    float a = Alpha[(size_t)e * heads + h];
    atomicAdd(&AGG[(size_t)d * C + c], Hm[(size_t)s * ldH + c] * a);
}

// ---------------- node epilogue: +bias, BN, ELU, write fp16 next-layer input ----------------
__global__ void k_post(const float* __restrict__ AGG, const float* __restrict__ b,
                       const float* __restrict__ g, const float* __restrict__ be,
                       const float* __restrict__ mu, const float* __restrict__ var,
                       half_t* __restrict__ Xout, int Nn, int C) {
    int i = blockIdx.x * blockDim.x + threadIdx.x;
    if (i >= Nn * C) return;
    int c = i % C;
    float x = AGG[i] + b[c];
    x = (x - mu[c]) * rsqrtf(var[c] + BN_EPS) * g[c] + be[c];
    x = x > 0.f ? x : (__expf(x) - 1.f);
    Xout[i] = (half_t)x;
}

// ---------------- final: +b3, log_softmax over 3 channels ----------------
__global__ void k_final(const float* __restrict__ AGG3, const float* __restrict__ b3,
                        float* __restrict__ out, int Nn) {
    int n = blockIdx.x * blockDim.x + threadIdx.x;
    if (n >= Nn) return;
    float v0 = AGG3[(size_t)n * 3 + 0] + b3[0];
    float v1 = AGG3[(size_t)n * 3 + 1] + b3[1];
    float v2 = AGG3[(size_t)n * 3 + 2] + b3[2];
    float m = fmaxf(v0, fmaxf(v1, v2));
    float l = __logf(__expf(v0 - m) + __expf(v1 - m) + __expf(v2 - m));
    out[(size_t)n * 3 + 0] = v0 - m - l;
    out[(size_t)n * 3 + 1] = v1 - m - l;
    out[(size_t)n * 3 + 2] = v2 - m - l;
}

// =====================================================================================
extern "C" void kernel_launch(void* const* d_in, const int* in_sizes, int n_in,
                              void* d_out, int out_size, void* d_ws, size_t ws_size,
                              hipStream_t stream) {
    const float* x   = (const float*)d_in[0];
    const int*   ei  = (const int*)  d_in[1];
    const float* W1  = (const float*)d_in[2];
    const float* a1s = (const float*)d_in[3];
    const float* a1d = (const float*)d_in[4];
    const float* b1  = (const float*)d_in[5];
    const float* g1  = (const float*)d_in[6];
    const float* be1 = (const float*)d_in[7];
    const float* m1  = (const float*)d_in[8];
    const float* v1  = (const float*)d_in[9];
    const float* W2  = (const float*)d_in[10];
    const float* a2s = (const float*)d_in[11];
    const float* a2d = (const float*)d_in[12];
    const float* b2  = (const float*)d_in[13];
    const float* g2  = (const float*)d_in[14];
    const float* be2 = (const float*)d_in[15];
    const float* m2  = (const float*)d_in[16];
    const float* v2  = (const float*)d_in[17];
    const float* W3  = (const float*)d_in[18];
    const float* a3s = (const float*)d_in[19];
    const float* a3d = (const float*)d_in[20];
    const float* b3  = (const float*)d_in[21];
    float* out = (float*)d_out;

    const int Nn = in_sizes[0] / 128;  // 50000
    const int E  = in_sizes[1] / 2;    // 850000
    const int* src = ei;
    const int* dst = ei + E;

    // ---- carve workspace (256B aligned slabs) ----
    char* wsp = (char*)d_ws;
    auto carve = [&](size_t bytes) -> void* {
        void* p = wsp;
        wsp += (bytes + 255) & ~(size_t)255;
        return p;
    };
    half_t* A16  = (half_t*)carve((size_t)Nn * 256 * sizeof(half_t)); // fp16 activations
    float*  H    = (float*) carve((size_t)Nn * 256 * sizeof(float));  // GEMM out
    float*  AGG  = (float*) carve((size_t)Nn * 256 * sizeof(float));  // aggregation out
    half_t* Wt1  = (half_t*)carve((size_t)256 * 128 * sizeof(half_t));
    half_t* Wt2  = (half_t*)carve((size_t)128 * 256 * sizeof(half_t));
    half_t* Wt3  = (half_t*)carve((size_t)16  * 128 * sizeof(half_t));
    float*  ALS  = (float*) carve((size_t)Nn * 4 * sizeof(float));
    float*  ALD  = (float*) carve((size_t)Nn * 4 * sizeof(float));
    float*  MX   = (float*) carve((size_t)Nn * 4 * sizeof(float));
    float*  SS   = (float*) carve((size_t)Nn * 4 * sizeof(float));
    float*  EBUF = (float*) carve((size_t)E  * 4 * sizeof(float));

    const int T = 256;
    auto blk = [](int n, int t) { return (n + t - 1) / t; };

    // ---- prep: fp16 input + transposed fp16 weights ----
    k_f2h   <<<blk(Nn * 128, T), T, 0, stream>>>(x, A16, Nn * 128);
    k_wtrans<<<blk(256 * 128, T), T, 0, stream>>>(W1, Wt1, 128, 256, 256);
    k_wtrans<<<blk(128 * 256, T), T, 0, stream>>>(W2, Wt2, 256, 128, 128);
    k_wtrans<<<blk(16  * 128, T), T, 0, stream>>>(W3, Wt3, 128, 3, 16);

    // =============== layer 1: 128 -> 4x64 (concat, D=256) ===============
    {
        const int K = 128, D = 256, HD = 4, F = 64;
        int waves = (Nn / 16) * ((D / 16) / 4);   // NB = 4
        k_wmma_gemm<4><<<blk(waves, 8), 256, 0, stream>>>(A16, Wt1, H, Nn, K, D);
        k_attn<<<blk(Nn * HD, T), T, 0, stream>>>(H, D, HD, F, a1s, a1d, ALS, ALD, Nn);
        k_fill<<<blk(Nn * HD, T), T, 0, stream>>>(MX, -3.0e38f, Nn * HD);
        k_fill<<<blk(Nn * HD, T), T, 0, stream>>>(SS, 0.0f, Nn * HD);
        k_fill<<<blk(Nn * D, T), T, 0, stream>>>(AGG, 0.0f, Nn * D);
        k_edge1<<<blk(E, T), T, 0, stream>>>(src, dst, ALS, ALD, EBUF, MX, E, HD);
        k_edge2<<<blk(E * HD, T), T, 0, stream>>>(dst, MX, EBUF, SS, E, HD);
        k_alpha<<<blk(E * HD, T), T, 0, stream>>>(dst, SS, EBUF, E, HD);
        int total = E * D;
        k_agg<<<blk(total, T), T, 0, stream>>>(src, dst, EBUF, H, D, AGG, D, F, HD, total);
        k_post<<<blk(Nn * D, T), T, 0, stream>>>(AGG, b1, g1, be1, m1, v1, A16, Nn, D);
    }

    // =============== layer 2: 256 -> 4x32 (concat, D=128) ===============
    {
        const int K = 256, D = 128, HD = 4, F = 32;
        int waves = (Nn / 16) * ((D / 16) / 4);   // NB = 4
        k_wmma_gemm<4><<<blk(waves, 8), 256, 0, stream>>>(A16, Wt2, H, Nn, K, D);
        k_attn<<<blk(Nn * HD, T), T, 0, stream>>>(H, D, HD, F, a2s, a2d, ALS, ALD, Nn);
        k_fill<<<blk(Nn * HD, T), T, 0, stream>>>(MX, -3.0e38f, Nn * HD);
        k_fill<<<blk(Nn * HD, T), T, 0, stream>>>(SS, 0.0f, Nn * HD);
        k_fill<<<blk(Nn * D, T), T, 0, stream>>>(AGG, 0.0f, Nn * D);
        k_edge1<<<blk(E, T), T, 0, stream>>>(src, dst, ALS, ALD, EBUF, MX, E, HD);
        k_edge2<<<blk(E * HD, T), T, 0, stream>>>(dst, MX, EBUF, SS, E, HD);
        k_alpha<<<blk(E * HD, T), T, 0, stream>>>(dst, SS, EBUF, E, HD);
        int total = E * D;
        k_agg<<<blk(total, T), T, 0, stream>>>(src, dst, EBUF, H, D, AGG, D, F, HD, total);
        k_post<<<blk(Nn * D, T), T, 0, stream>>>(AGG, b2, g2, be2, m2, v2, A16, Nn, D);
    }

    // =============== layer 3: 128 -> 1x3 (mean over 1 head == identity) ===============
    {
        const int K = 128, Dp = 16, C = 3, HD = 1, F = 3;
        int waves = (Nn / 16) * (Dp / 16);        // NB = 1
        k_wmma_gemm<1><<<blk(waves, 8), 256, 0, stream>>>(A16, Wt3, H, Nn, K, Dp);
        k_attn<<<blk(Nn * HD, T), T, 0, stream>>>(H, Dp, HD, F, a3s, a3d, ALS, ALD, Nn);
        k_fill<<<blk(Nn * HD, T), T, 0, stream>>>(MX, -3.0e38f, Nn * HD);
        k_fill<<<blk(Nn * HD, T), T, 0, stream>>>(SS, 0.0f, Nn * HD);
        k_fill<<<blk(Nn * C, T), T, 0, stream>>>(AGG, 0.0f, Nn * C);
        k_edge1<<<blk(E, T), T, 0, stream>>>(src, dst, ALS, ALD, EBUF, MX, E, HD);
        k_edge2<<<blk(E * HD, T), T, 0, stream>>>(dst, MX, EBUF, SS, E, HD);
        k_alpha<<<blk(E * HD, T), T, 0, stream>>>(dst, SS, EBUF, E, HD);
        int total = E * C;
        k_agg<<<blk(total, T), T, 0, stream>>>(src, dst, EBUF, H, Dp, AGG, C, F, HD, total);
        k_final<<<blk(Nn, T), T, 0, stream>>>(AGG, b3, out, Nn);
    }
}
